// WeightedMultiHeadGAT_14757507629162
// MI455X (gfx1250) — matrix operand order repaired and checked by reading, not compiled, not run on hardware
//
#include <hip/hip_runtime.h>
#include <hip/hip_bf16.h>

typedef __attribute__((ext_vector_type(2)))  float  v2f;
typedef __attribute__((ext_vector_type(8)))  float  v8f;
typedef __attribute__((ext_vector_type(16))) __bf16 v16bf;
typedef __attribute__((ext_vector_type(2)))  int    v2i;
typedef __attribute__((ext_vector_type(4)))  int    v4i;

#define N_NODES 4096
#define IN_F    256
#define HEADS   8
#define DHEAD   32
#define HD      256
#define ALPHA   0.2f

#define AS1 __attribute__((address_space(1)))
#define AS3 __attribute__((address_space(3)))

// ---------------------------------------------------------------------------
// gfx1250 async global->LDS copies (ASYNCcnt-tracked). Builtins are declared
// in this toolchain (probe-confirmed); params are typed (int / v2i / v4i)
// with (global_src, lds_dst, imm_offset, cpol) ordering.
// ---------------------------------------------------------------------------
__device__ __forceinline__ void async_b32(void* lds, const void* g) {
#if __has_builtin(__builtin_amdgcn_global_load_async_to_lds_b32)
  __builtin_amdgcn_global_load_async_to_lds_b32((AS1 int*)g, (AS3 int*)lds, 0, 0);
#else
  asm volatile("global_load_async_to_lds_b32 %0, %1, off"
               :: "v"((unsigned)(uintptr_t)lds),
                  "v"((unsigned long long)(uintptr_t)g) : "memory");
#endif
}
__device__ __forceinline__ void async_b64(void* lds, const void* g) {
#if __has_builtin(__builtin_amdgcn_global_load_async_to_lds_b64)
  __builtin_amdgcn_global_load_async_to_lds_b64((AS1 v2i*)g, (AS3 v2i*)lds, 0, 0);
#else
  asm volatile("global_load_async_to_lds_b64 %0, %1, off"
               :: "v"((unsigned)(uintptr_t)lds),
                  "v"((unsigned long long)(uintptr_t)g) : "memory");
#endif
}
__device__ __forceinline__ void async_b128(void* lds, const void* g) {
#if __has_builtin(__builtin_amdgcn_global_load_async_to_lds_b128)
  __builtin_amdgcn_global_load_async_to_lds_b128((AS1 v4i*)g, (AS3 v4i*)lds, 0, 0);
#else
  asm volatile("global_load_async_to_lds_b128 %0, %1, off"
               :: "v"((unsigned)(uintptr_t)lds),
                  "v"((unsigned long long)(uintptr_t)g) : "memory");
#endif
}
__device__ __forceinline__ void wait_async0() {
#if __has_builtin(__builtin_amdgcn_s_wait_asynccnt)
  __builtin_amdgcn_s_wait_asynccnt(0);
#else
  asm volatile("s_wait_asynccnt 0x0" ::: "memory");
#endif
}

// ---------------------------------------------------------------------------
// Kernel 1: projection GEMMs in full f32 via V_WMMA_F32_16X16X4_F32.
//   which=0: h = x @ W^T   -> hbuf (f32) and hbft (bf16, transposed [ch][node])
//   which=1: res = x @ Wr^T + br -> out (overwrites d_out; attn kernel adds)
// ---------------------------------------------------------------------------
__global__ __launch_bounds__(128) void gat_proj(
    const float* __restrict__ x, const float* __restrict__ W,
    const float* __restrict__ Wr, const float* __restrict__ br,
    float* __restrict__ hbuf, __bf16* __restrict__ hbft,
    float* __restrict__ out)
{
  int wid  = blockIdx.x * 4 + (threadIdx.x >> 5);
  int lane = threadIdx.x & 31;
  int which = wid >> 12;          // 8192 waves: 4096 for h, 4096 for res
  int t  = wid & 4095;
  int m0 = (t >> 4) * 16;
  int n0 = (t & 15) * 16;
  const float* Bmat = which ? Wr : W;

  int mn = lane & 15;             // row for A, col for B
  int kb = (lane >> 4) * 2;       // K sub-group (f32 16x4 layout)
  const float* arow = x    + (size_t)(m0 + mn) * IN_F;
  const float* brow = Bmat + (size_t)(n0 + mn) * IN_F;   // B(k,n) = W[n,k]

  v8f acc = {};
#pragma unroll 4
  for (int k0 = 0; k0 < IN_F; k0 += 4) {
    v2f a, b;
    a.x = arow[k0 + kb]; a.y = arow[k0 + kb + 1];
    b.x = brow[k0 + kb]; b.y = brow[k0 + kb + 1];
    acc = __builtin_amdgcn_wmma_f32_16x16x4_f32(
        false, a, false, b, (short)0, acc, false, false);
  }

  int hi  = lane >> 4;
  int col = n0 + mn;
  if (which) {                    // wave-uniform branch
    float bias = br[col];
#pragma unroll
    for (int r = 0; r < 8; ++r) {
      int row = m0 + r + 8 * hi;  // C/D layout: M = r + 8*(lane>=16)
      out[(size_t)row * HD + col] = acc[r] + bias;
    }
  } else {
#pragma unroll
    for (int r = 0; r < 8; ++r) {
      int row = m0 + r + 8 * hi;
      hbuf[(size_t)row * HD + col]      = acc[r];
      hbft[(size_t)col * N_NODES + row] = (__bf16)acc[r];  // transposed bf16 V
    }
  }
}

// ---------------------------------------------------------------------------
// Kernel 2: s_src[n,h] = <h[n,h,:], a_src[h,:]>, same for s_dst.
// ---------------------------------------------------------------------------
__global__ __launch_bounds__(256) void gat_scores(
    const float* __restrict__ h, const float* __restrict__ a_src,
    const float* __restrict__ a_dst, float* __restrict__ ssrc,
    float* __restrict__ sdst)
{
  int idx = blockIdx.x * blockDim.x + threadIdx.x;
  if (idx >= N_NODES * HEADS) return;
  int n = idx >> 3, hd = idx & 7;
  const float* hv = h + (size_t)n * HD + hd * DHEAD;
  const float* as = a_src + hd * DHEAD;
  const float* ad = a_dst + hd * DHEAD;
  float s1 = 0.f, s2 = 0.f;
#pragma unroll
  for (int d = 0; d < DHEAD; ++d) { float v = hv[d]; s1 += v * as[d]; s2 += v * ad[d]; }
  ssrc[idx] = s1;
  sdst[idx] = s2;
}

// ---------------------------------------------------------------------------
// Kernel 3: softmax pass 1 — per (row i, head): online max/sum over j.
// ---------------------------------------------------------------------------
__global__ __launch_bounds__(256) void gat_rowstats(
    const float* __restrict__ adj, const float* __restrict__ ssrc,
    const float* __restrict__ sdst, const float* __restrict__ We,
    const float* __restrict__ be, float* __restrict__ rowmax,
    float* __restrict__ rowsum)
{
  int i    = blockIdx.x;
  int head = threadIdx.x >> 5;
  int lane = threadIdx.x & 31;
  float weh = We[head], beh = be[head];
  float si  = ssrc[i * HEADS + head];
  const float* arow = adj + (size_t)i * N_NODES;

  float m = -INFINITY, s = 0.f;
  for (int j = lane; j < N_NODES; j += 32) {
    float a = arow[j];
    if (a > 0.f) {
      float sc = si + sdst[j * HEADS + head] + a * weh + beh;
      sc = sc > 0.f ? sc : ALPHA * sc;
      float nm = fmaxf(m, sc);
      s = s * __expf(m - nm) + __expf(sc - nm);   // m=-inf -> s==0, 0*0 ok
      m = nm;
    }
  }
#pragma unroll
  for (int off = 16; off > 0; off >>= 1) {
    float om = __shfl_xor(m, off, 32);
    float os = __shfl_xor(s, off, 32);
    float nm = fmaxf(m, om);
    float e1 = (m  > -1e37f) ? __expf(m  - nm) : 0.f;
    float e2 = (om > -1e37f) ? __expf(om - nm) : 0.f;
    s = s * e1 + os * e2;
    m = nm;
  }
  if (lane == 0) {
    rowmax[i * HEADS + head] = m;
    rowsum[i * HEADS + head] = s;
  }
}

// ---------------------------------------------------------------------------
// Kernel 4: softmax pass 2 + P·V via V_WMMA_F32_16X16X32_BF16.
// One block = 16-row i-tile; 8 waves = 8 heads. adj / s_dst / V(bf16,
// transposed) tiles are staged with async global->LDS copies, ping-pong
// double buffered so staging of tile t+1 overlaps compute of tile t.
// ---------------------------------------------------------------------------
__global__ __launch_bounds__(256) void gat_attn(
    const float* __restrict__ adj, const __bf16* __restrict__ hbft,
    const float* __restrict__ ssrc, const float* __restrict__ sdst,
    const float* __restrict__ We, const float* __restrict__ be,
    const float* __restrict__ rowmax, const float* __restrict__ rowsum,
    float* __restrict__ out)
{
  __shared__ float  adjT[2][16 * 32];     // adj tile [i-row][j]
  __shared__ float  sdT [2][32 * 8];      // s_dst tile [j][head]
  __shared__ __bf16 hTt [2][256 * 32];    // V tile transposed [ch][j]

  int i0   = blockIdx.x * 16;
  int tid  = threadIdx.x;
  int head = tid >> 5;
  int lane = tid & 31;
  int m    = lane & 15;                   // A row / B,C column
  int hi   = lane >> 4;

  float weh = We[head], beh = be[head];
  int rowm  = i0 + m;
  float cs  = ssrc  [rowm * HEADS + head];
  float rm  = rowmax[rowm * HEADS + head];

  // per-thread staging assignments
  int ae = tid * 2;                       // adj: 2 contiguous floats / thread
  int ar = ae >> 5, ac = ae & 31;
  const float*  adjg = adj  + (size_t)(i0 + ar) * N_NODES + ac;
  const __bf16* hg   = hbft + (size_t)tid * N_NODES;   // channel = tid

  v8f acc0 = {}, acc1 = {};

  // prologue: stage tile 0 into buffer 0
  {
    async_b64 (&adjT[0][ae], adjg);
    async_b32 (&sdT[0][tid], sdst + 0 * HEADS + tid);
    __bf16* l = &hTt[0][tid * 32];
    async_b128(l,      hg);
    async_b128(l + 8,  hg + 8);
    async_b128(l + 16, hg + 16);
    async_b128(l + 24, hg + 24);
  }

  for (int t = 0; t < N_NODES / 32; ++t) {
    int j0  = t * 32;
    int buf = t & 1;
    wait_async0();                 // this wave's staging for buf complete
    __syncthreads();               // all waves' staging complete

    if (t + 1 < N_NODES / 32) {    // stage next tile into other buffer
      int nb = buf ^ 1, jn = j0 + 32;
      async_b64 (&adjT[nb][ae], adjg + jn);
      async_b32 (&sdT[nb][tid], sdst + jn * HEADS + tid);
      __bf16* l = &hTt[nb][tid * 32];
      const __bf16* g = hg + jn;
      async_b128(l,      g);
      async_b128(l + 8,  g + 8);
      async_b128(l + 16, g + 16);
      async_b128(l + 24, g + 24);
    }

    // A fragment = P tile (16 rows x 32 j), bf16.
    // 16-bit A 16x32 layout (matches B): elt e ->
    //   k = e + 16*(lane>=16)
    v16bf af;
#pragma unroll
    for (int e = 0; e < 16; ++e) {
      int r = e >> 1, halfk = e & 1;
      int k = e + (hi << 4);
      float a = adjT[buf][m * 32 + k];
      float p = 0.f;
      if (a > 0.f) {
        float sc = cs + sdT[buf][k * 8 + head] + a * weh + beh;
        sc = sc > 0.f ? sc : ALPHA * sc;
        p = __expf(sc - rm);       // <= 1, safe in bf16
      }
      af[e] = (__bf16)p;
    }

    // B fragments: lanes 0-15 hold K 0-15, lanes 16-31 hold K 16-31;
    // transposed V tile makes each fragment one contiguous 32B LDS read.
    v16bf b0 = *(const v16bf*)&hTt[buf][(head * DHEAD + m)      * 32 + hi * 16];
    v16bf b1 = *(const v16bf*)&hTt[buf][(head * DHEAD + 16 + m) * 32 + hi * 16];

    acc0 = __builtin_amdgcn_wmma_f32_16x16x32_bf16(
        false, af, false, b0, (short)0, acc0, false, false);
    acc1 = __builtin_amdgcn_wmma_f32_16x16x32_bf16(
        false, af, false, b1, (short)0, acc1, false, false);

    __syncthreads();               // everyone done reading buf before overwrite
  }

  // epilogue: normalize by rowsum, accumulate onto residual in d_out
#pragma unroll
  for (int r = 0; r < 8; ++r) {
    int row = i0 + r + 8 * hi;
    float inv = 1.0f / rowsum[row * HEADS + head];
    size_t base = (size_t)row * HD + head * DHEAD;
    out[base + m]      += acc0[r] * inv;
    out[base + 16 + m] += acc1[r] * inv;
  }
}

// ---------------------------------------------------------------------------
extern "C" void kernel_launch(void* const* d_in, const int* in_sizes, int n_in,
                              void* d_out, int out_size, void* d_ws, size_t ws_size,
                              hipStream_t stream) {
  const float* x     = (const float*)d_in[0];
  const float* adj   = (const float*)d_in[1];
  const float* W     = (const float*)d_in[2];
  const float* We    = (const float*)d_in[3];
  const float* be    = (const float*)d_in[4];
  const float* a_src = (const float*)d_in[5];
  const float* a_dst = (const float*)d_in[6];
  const float* Wr    = (const float*)d_in[7];
  const float* br    = (const float*)d_in[8];
  float* out = (float*)d_out;

  char* ws = (char*)d_ws;
  float*  hbuf   = (float*)ws;                          // 4 MB: h [N,256] f32
  __bf16* hbft   = (__bf16*)(ws + ((size_t)4 << 20));   // 2 MB: h^T [256,N] bf16
  float*  ssrc   = (float*)(ws + ((size_t)6 << 20));    // [N,H]
  float*  sdst   = ssrc   + N_NODES * HEADS;
  float*  rowmax = sdst   + N_NODES * HEADS;
  float*  rowsum = rowmax + N_NODES * HEADS;

  hipLaunchKernelGGL(gat_proj,     dim3(2048), dim3(128), 0, stream,
                     x, W, Wr, br, hbuf, hbft, out);
  hipLaunchKernelGGL(gat_scores,   dim3((N_NODES * HEADS + 255) / 256), dim3(256), 0, stream,
                     hbuf, a_src, a_dst, ssrc, sdst);
  hipLaunchKernelGGL(gat_rowstats, dim3(N_NODES), dim3(256), 0, stream,
                     adj, ssrc, sdst, We, be, rowmax, rowsum);
  hipLaunchKernelGGL(gat_attn,     dim3(N_NODES / 16), dim3(256), 0, stream,
                     adj, hbft, ssrc, sdst, We, be, rowmax, rowsum, out);
}
